// MultiHeadAttention_19121194402429
// MI455X (gfx1250) — matrix operand order repaired and checked by reading, not compiled, not run on hardware
//
#include <hip/hip_runtime.h>

typedef __attribute__((ext_vector_type(16))) _Float16 v16h;
typedef __attribute__((ext_vector_type(8)))  _Float16 v8h;
typedef __attribute__((ext_vector_type(8)))  float    v8f;

#define BB  16
#define SS  2048
#define DIN 1024
#define DD  256

#ifndef __has_builtin
#define __has_builtin(x) 0
#endif

#if __has_builtin(__builtin_amdgcn_global_load_async_to_lds_b128)
#define HAVE_ASYNC 1
typedef int v4i __attribute__((vector_size(16)));
typedef __attribute__((address_space(1))) v4i gv4i_t;   // global int4
typedef __attribute__((address_space(3))) v4i lv4i_t;   // LDS int4
#else
#define HAVE_ASYNC 0
#endif

static __device__ __forceinline__ void wait_async_le(int n) {
#if HAVE_ASYNC
#if __has_builtin(__builtin_amdgcn_s_wait_asynccnt)
  if (n == 0) __builtin_amdgcn_s_wait_asynccnt(0);
  else        __builtin_amdgcn_s_wait_asynccnt(16);
#else
  if (n == 0) asm volatile("s_wait_asynccnt 0x0" ::: "memory");
  else        asm volatile("s_wait_asynccnt 0x10" ::: "memory");
#endif
#endif
  (void)n;
}

// ---------- WMMA fragment helpers (layouts per CDNA5 ISA 7.12.2) ----------

// A fragment 16x32 f16: lane l holds row m=l%16; elements j=0..7 at
// K = (l/16)*8 + j, elements j=8..15 at K = 16 + (l/16)*8 + (j-8).
static __device__ __forceinline__ v16h load_a_frag(const _Float16* p, int ld) {
  const int lane = threadIdx.x & 31;
  const int m  = lane & 15;
  const int kb = (lane >> 4) << 3;
  const _Float16* r = p + m * ld + kb;
  v8h lo = *(const v8h*)(r);
  v8h hi = *(const v8h*)(r + 16);
  return __builtin_shufflevector(lo, hi, 0,1,2,3,4,5,6,7,8,9,10,11,12,13,14,15);
}

// B fragment 32x16 f16 from a TRANSPOSED source Bt[n][k] (k contiguous):
// lane l holds column n=l%16; elements j=0..7 at K = (l/16)*8 + j,
// elements j=8..15 at K = 16 + (l/16)*8 + (j-8).
static __device__ __forceinline__ v16h load_b_frag_t(const _Float16* pt, int ld) {
  const int lane = threadIdx.x & 31;
  const int n  = lane & 15;
  const int kb = (lane >> 4) << 3;
  const _Float16* r = pt + n * ld + kb;
  v8h lo = *(const v8h*)(r);
  v8h hi = *(const v8h*)(r + 16);
  return __builtin_shufflevector(lo, hi, 0,1,2,3,4,5,6,7,8,9,10,11,12,13,14,15);
}

static __device__ __forceinline__ v8f wmma_f16(v16h a, v16h b, v8f c) {
  return __builtin_amdgcn_wmma_f32_16x16x32_f16(
      /*neg_a=*/false, a, /*neg_b=*/false, b,
      /*c_mod=*/(short)0, c, /*reuse_a=*/false, /*reuse_b=*/false);
}

// ---------- Kernel 0a: x fp32 -> f16 ----------
__global__ __launch_bounds__(256) void cvt_x(const float* __restrict__ x,
                                             _Float16* __restrict__ xh, int n) {
  int i = (blockIdx.x * 256 + threadIdx.x) * 8;
  if (i + 8 <= n) {
    float4 a = *(const float4*)(x + i);
    float4 b = *(const float4*)(x + i + 4);
    v8h o;
    o[0] = (_Float16)a.x; o[1] = (_Float16)a.y;
    o[2] = (_Float16)a.z; o[3] = (_Float16)a.w;
    o[4] = (_Float16)b.x; o[5] = (_Float16)b.y;
    o[6] = (_Float16)b.z; o[7] = (_Float16)b.w;
    *(v8h*)(xh + i) = o;
  }
}

// ---------- Kernel 0b: w [K=1024][N=256] fp32 -> wT [3][N=256][K=1024] f16 ----
__global__ __launch_bounds__(256) void cvt_wT(const float* __restrict__ wq,
                                              const float* __restrict__ wk,
                                              const float* __restrict__ wv,
                                              _Float16* __restrict__ wT) {
  int idx = blockIdx.x * 256 + threadIdx.x;     // over 3*256*1024 (pow2 -> shifts)
  int m = idx >> 18;
  int r = idx & ((DD * DIN) - 1);
  int n = r >> 10;
  int k = r & (DIN - 1);
  const float* w = (m == 0) ? wq : ((m == 1) ? wk : wv);
  wT[idx] = (_Float16)w[k * DD + n];
}

// ---------- Kernel 1: QKV projection GEMM (ping-pong pipelined) ----------
// grid = (3 matrices, 2048 row-tiles): the 3 blocks sharing a row tile are
// adjacent in dispatch order so the x slice is served from cache after the
// first matrix reads it.
__global__ __launch_bounds__(128) void qkv_gemm(const _Float16* __restrict__ xh,
                                                const _Float16* __restrict__ wT,
                                                _Float16* __restrict__ Qh,
                                                _Float16* __restrict__ Kh,
                                                _Float16* __restrict__ VhT) {
  const int mat  = blockIdx.x;
  const int mt   = blockIdx.y;
  const int wave = threadIdx.x >> 5;
  const int lane = threadIdx.x & 31;
  const int row0 = mt * 16;
  const _Float16* wTm = wT + (size_t)mat * DD * DIN;
  const _Float16* xrow = xh + (size_t)row0 * DIN;
  const size_t nbase = (size_t)(wave * 64) * DIN;

  v8f acc[4] = {};
  // set A: fragments for k = 0
  v16h aA = load_a_frag(xrow, DIN);
  v16h bA[4];
#pragma unroll
  for (int i = 0; i < 4; ++i)
    bA[i] = load_b_frag_t(wTm + nbase + (size_t)(i * 16) * DIN, DIN);

  for (int kk = 0; kk < DIN; kk += 64) {
    // set B: fragments for kk+32 (loads overlap set-A WMMAs)
    v16h aB = load_a_frag(xrow + kk + 32, DIN);
    v16h bB[4];
#pragma unroll
    for (int i = 0; i < 4; ++i)
      bB[i] = load_b_frag_t(wTm + nbase + (size_t)(i * 16) * DIN + kk + 32, DIN);
#pragma unroll
    for (int i = 0; i < 4; ++i) acc[i] = wmma_f16(aA, bA[i], acc[i]);
    if (kk + 64 < DIN) {   // set A: fragments for kk+64 (overlap set-B WMMAs)
      aA = load_a_frag(xrow + kk + 64, DIN);
#pragma unroll
      for (int i = 0; i < 4; ++i)
        bA[i] = load_b_frag_t(wTm + nbase + (size_t)(i * 16) * DIN + kk + 64, DIN);
    }
#pragma unroll
    for (int i = 0; i < 4; ++i) acc[i] = wmma_f16(aB, bB[i], acc[i]);
  }

  const int nin = lane & 15;
  const int mb  = (lane >> 4) * 8;
#pragma unroll
  for (int i = 0; i < 4; ++i) {
    const int n0 = wave * 64 + i * 16 + nin;
#pragma unroll
    for (int v = 0; v < 8; ++v) {
      const int row = row0 + mb + v;
      const float val = acc[i][v];
      if (mat == 0) {
        Qh[(size_t)row * DD + n0] = (_Float16)(val * 0.0625f);  // fold 1/sqrt(D)
      } else if (mat == 1) {
        Kh[(size_t)row * DD + n0] = (_Float16)val;
      } else {
        const int b = row >> 11;            // S = 2048
        const int s = row & (SS - 1);
        VhT[(size_t)b * DD * SS + (size_t)n0 * SS + s] = (_Float16)val;
      }
    }
  }
}

// ---------- cooperative K/V tile staging into LDS ----------
// K tile: 32 rows x 256 f16 (contiguous 16KB).  V tile: 256 rows x 32 f16.
// 128 threads, 16 x b128 async copies per thread per tile pair.
static __device__ __forceinline__ void stage_tiles(const _Float16* __restrict__ Kb,
                                                   const _Float16* __restrict__ Vb,
                                                   _Float16* kdst, _Float16* vdst,
                                                   int t0, int tid) {
  const _Float16* ksrc = Kb + (size_t)t0 * DD;
#if HAVE_ASYNC
#pragma unroll
  for (int c = 0; c < 8; ++c) {
    const int off = (c * 128 + tid) * 8;            // 16B chunks
    __builtin_amdgcn_global_load_async_to_lds_b128(
        (gv4i_t*)(ksrc + off), (lv4i_t*)(kdst + off), 0, 0);
  }
#pragma unroll
  for (int i = 0; i < 2; ++i) {
    const int rrow = tid * 2 + i;                    // 0..255 (d index)
    const _Float16* vsrc = Vb + (size_t)rrow * SS + t0;
#pragma unroll
    for (int c = 0; c < 4; ++c) {
      __builtin_amdgcn_global_load_async_to_lds_b128(
          (gv4i_t*)(vsrc + c * 8), (lv4i_t*)(vdst + rrow * 32 + c * 8), 0, 0);
    }
  }
#else
#pragma unroll
  for (int c = 0; c < 8; ++c) {
    const int off = (c * 128 + tid) * 8;
    *(v8h*)(kdst + off) = *(const v8h*)(ksrc + off);
  }
#pragma unroll
  for (int i = 0; i < 2; ++i) {
    const int rrow = tid * 2 + i;
    const _Float16* vsrc = Vb + (size_t)rrow * SS + t0;
#pragma unroll
    for (int c = 0; c < 4; ++c)
      *(v8h*)(vdst + rrow * 32 + c * 8) = *(const v8h*)(vsrc + c * 8);
  }
#endif
}

// ---------- Kernel 2: fused flash attention ----------
// 4 waves/block (same batch b), one 16-query tile per wave. K/V tiles staged
// in LDS (double-buffered async copies), online softmax over 32-key chunks.
__global__ __launch_bounds__(128) void attn(const _Float16* __restrict__ Qh,
                                            const _Float16* __restrict__ Kh,
                                            const _Float16* __restrict__ VhT,
                                            float* __restrict__ out) {
  __shared__ _Float16 Klds[2][32 * 256];   // 2 x 16KB
  __shared__ _Float16 Vlds[2][256 * 32];   // 2 x 16KB
  __shared__ _Float16 Plds[4][16 * 32];    // 4 x 1KB wave-private P staging

  const int tid  = threadIdx.x;
  const int wave = tid >> 5;
  const int lane = tid & 31;
  const int tile = blockIdx.x * 4 + wave;  // 0..2047
  const int b    = tile >> 7;              // all 4 waves share b (128 tiles/b)
  const int q0   = (tile & 127) * 16;
  const _Float16* Qb = Qh + ((size_t)b * SS + q0) * DD;
  const _Float16* Kb = Kh + (size_t)b * SS * DD;
  const _Float16* Vb = VhT + (size_t)b * DD * SS;
  _Float16* myP = &Plds[wave][0];

  // Preload Q fragments (16 x 256 = 8 chunks of K=32), already scaled.
  v16h qa[8];
#pragma unroll
  for (int dk = 0; dk < 8; ++dk) qa[dk] = load_a_frag(Qb + dk * 32, DD);

  v8f o[16] = {};                 // 16x256 fp32 output accumulator
  float mrow[8], lrow[8];
#pragma unroll
  for (int v = 0; v < 8; ++v) { mrow[v] = -3.0e38f; lrow[v] = 0.0f; }

  stage_tiles(Kb, Vb, &Klds[0][0], &Vlds[0][0], 0, tid);   // prologue

  const int NIT = SS / 32;
  for (int it = 0; it < NIT; ++it) {
    const int buf = it & 1;
    if (it + 1 < NIT)
      stage_tiles(Kb, Vb, &Klds[buf ^ 1][0], &Vlds[buf ^ 1][0], (it + 1) * 32, tid);
    wait_async_le((it + 1 < NIT) ? 16 : 0);   // current buf's copies complete
    __syncthreads();                          // ...across all waves

    const _Float16* Kt = &Klds[buf][0];
    const _Float16* Vt = &Vlds[buf][0];

    // ---- scores S[16x32] = Q(16x256) * K^T, via 2 N-tiles x 8 K-chunks ----
    v8f s0 = {}, s1 = {};
#pragma unroll
    for (int dk = 0; dk < 8; ++dk) {
      v16h k0f = load_b_frag_t(Kt + dk * 32, DD);
      v16h k1f = load_b_frag_t(Kt + 16 * DD + dk * 32, DD);
      s0 = wmma_f16(qa[dk], k0f, s0);
      s1 = wmma_f16(qa[dk], k1f, s1);
    }
    // ---- online softmax (row = VGPR v + 8*(lane/16); 16-lane reductions) ----
#pragma unroll
    for (int v = 0; v < 8; ++v) {
      float x = fmaxf(s0[v], s1[v]);
      x = fmaxf(x, __shfl_xor(x, 1, 32));
      x = fmaxf(x, __shfl_xor(x, 2, 32));
      x = fmaxf(x, __shfl_xor(x, 4, 32));
      x = fmaxf(x, __shfl_xor(x, 8, 32));
      const float mnew = fmaxf(mrow[v], x);
      const float corr = __expf(mrow[v] - mnew);
      const float p0 = __expf(s0[v] - mnew);
      const float p1 = __expf(s1[v] - mnew);
      s0[v] = p0; s1[v] = p1;
      float rs = p0 + p1;
      rs += __shfl_xor(rs, 1, 32);
      rs += __shfl_xor(rs, 2, 32);
      rs += __shfl_xor(rs, 4, 32);
      rs += __shfl_xor(rs, 8, 32);
      lrow[v] = lrow[v] * corr + rs;
      mrow[v] = mnew;
#pragma unroll
      for (int dt = 0; dt < 16; ++dt) o[dt][v] *= corr;   // rescale O
    }
    // ---- re-layout P: C-layout VGPRs -> row-major [16][32] f16 in LDS ----
    {
      const int col = lane & 15;
      const int mb  = (lane >> 4) * 8;
#pragma unroll
      for (int v = 0; v < 8; ++v) {
        myP[(mb + v) * 32 + col]      = (_Float16)s0[v];
        myP[(mb + v) * 32 + col + 16] = (_Float16)s1[v];
      }
    }
    v16h pa = load_a_frag(myP, 32);
    // ---- O += P(16x32) * V(32x256): 16 d-tiles from staged transposed V ----
#pragma unroll
    for (int dt = 0; dt < 16; ++dt) {
      v16h vf = load_b_frag_t(Vt + dt * 16 * 32, 32);
      o[dt] = wmma_f16(pa, vf, o[dt]);
    }
    __syncthreads();   // all waves done with buf before it is overwritten
  }

  // ---- epilogue: normalize by row sums, store fp32 ----
  const int n  = lane & 15;
  const int mb = (lane >> 4) * 8;
#pragma unroll
  for (int v = 0; v < 8; ++v) {
    const float inv = 1.0f / lrow[v];
    const int srow = q0 + mb + v;
    float* orow = out + ((size_t)b * SS + srow) * DD;
#pragma unroll
    for (int dt = 0; dt < 16; ++dt) orow[dt * 16 + n] = o[dt][v] * inv;
  }
}

// ---------- launcher ----------
extern "C" void kernel_launch(void* const* d_in, const int* in_sizes, int n_in,
                              void* d_out, int out_size, void* d_ws, size_t ws_size,
                              hipStream_t stream) {
  const float* x  = (const float*)d_in[0];
  const float* wq = (const float*)d_in[1];
  const float* wk = (const float*)d_in[2];
  const float* wv = (const float*)d_in[3];
  float* out = (float*)d_out;

  char* ws = (char*)d_ws;
  _Float16* xh  = (_Float16*)(ws);                               // 64 MB
  _Float16* wT  = (_Float16*)(ws + (size_t)67108864);            // 1.5 MB
  _Float16* Qh  = (_Float16*)(ws + (size_t)68681728);            // 16 MB
  _Float16* Kh  = (_Float16*)(ws + (size_t)85458944);            // 16 MB
  _Float16* VhT = (_Float16*)(ws + (size_t)102236160);           // 16 MB

  const int nx = BB * SS * DIN;                                  // 33554432
  cvt_x<<<nx / (256 * 8), 256, 0, stream>>>(x, xh, nx);
  cvt_wT<<<(3 * DD * DIN) / 256, 256, 0, stream>>>(wq, wk, wv, wT);

  dim3 g1(3, BB * SS / 16, 1);
  qkv_gemm<<<g1, 128, 0, stream>>>(xh, wT, Qh, Kh, VhT);

  attn<<<(BB * SS / 16) / 4, 128, 0, stream>>>(Qh, Kh, VhT, out);
}
